// Gate_29334626632566
// MI455X (gfx1250) — matrix operand order repaired and checked by reading, not compiled, not run on hardware
//
#include <hip/hip_runtime.h>
#include <math.h>

// CDNA5 / gfx1250 MoE gate: scores = x @ W^T (bf16-split WMMA), sigmoid,
// top-8 per token, normalized weights + indices.
//
// x: [16384, 4096] f32, W: [64, 4096] f32
// d_out: [16384*8] weights (f32) followed by [16384*8] indices (as f32).
//
// Strategy: W is tiny (1 MB) and shared by everyone -> pre-convert it once
// per launch into d_ws as bf16 hi + bf16 residual lo planes. The GEMM kernel
// then only converts the streamed A (x) fragments, keeping the K-loop near
// 12 WMMAs + 20 b128 loads + ~32 VALU per chunk (A conversion co-executes
// with the TRANS-class 16-bit WMMAs).

typedef __attribute__((ext_vector_type(16))) __bf16 v16bf;
typedef __attribute__((ext_vector_type(8)))  float  v8f;

#define TOKENS 16384
#define DIM    4096
#define NEXP   64
#define TOPK   8
#define TOK_PER_BLOCK 128   // 8 waves * 16 tokens

static __device__ __forceinline__ unsigned short bf16_bits(__bf16 h) {
    union { __bf16 b; unsigned short u; } cv; cv.b = h; return cv.u;
}

// ---- one-shot weight pre-conversion: f32 -> bf16 hi + bf16 residual lo ----
// 64*4096 = 262144 elems; 65536 threads x 4 elems (float4 in, uint2 out).
__global__ __launch_bounds__(256) void conv_weight(
    const float* __restrict__ w,
    unsigned int* __restrict__ whi,     // [NEXP*DIM/2] packed bf16 pairs
    unsigned int* __restrict__ wlo)
{
    const int gid = blockIdx.x * 256 + threadIdx.x;
    const float4 v = ((const float4*)w)[gid];
    float f[4] = {v.x, v.y, v.z, v.w};
    unsigned int hi[2], lo[2];
    #pragma unroll
    for (int p = 0; p < 2; ++p) {
        __bf16 h0 = (__bf16)f[2*p + 0];
        __bf16 h1 = (__bf16)f[2*p + 1];
        __bf16 l0 = (__bf16)(f[2*p + 0] - (float)h0);
        __bf16 l1 = (__bf16)(f[2*p + 1] - (float)h1);
        hi[p] = (unsigned int)bf16_bits(h0) | ((unsigned int)bf16_bits(h1) << 16);
        lo[p] = (unsigned int)bf16_bits(l0) | ((unsigned int)bf16_bits(l1) << 16);
    }
    ((uint2*)whi)[gid] = make_uint2(hi[0], hi[1]);
    ((uint2*)wlo)[gid] = make_uint2(lo[0], lo[1]);
}

// ---- main kernel: GEMM (bf16-split WMMA) + sigmoid + top-8 ----
template <bool PRECONV>
__global__ __launch_bounds__(256) void moe_gate_wmma(
    const float* __restrict__ x,
    const float* __restrict__ w,
    const v16bf* __restrict__ whi,      // pre-converted B planes (PRECONV only)
    const v16bf* __restrict__ wlo,
    float* __restrict__ out)
{
    __shared__ float sc[TOK_PER_BLOCK][NEXP + 1];   // padded vs 64 banks

    const int tid  = threadIdx.x;
    const int wave = tid >> 5;        // 0..7
    const int lane = tid & 31;
    const int half = lane >> 4;       // K-group select
    const int col  = lane & 15;       // token row (A) / expert col (B) / N col (C)
    const int tokBase = blockIdx.x * TOK_PER_BLOCK;
    const int myTok   = tokBase + wave * 16 + col;

    v8f acc[4] = {};                  // 16 tokens x 64 experts

    // A-fragment base: lanes<16 start at K+0, lanes>=16 at K+16 (contiguous K, matches B)
    const float* pa = x + (size_t)myTok * DIM + half * 16;

    for (int kb = 0; kb < DIM; kb += 32) {
        __builtin_prefetch(pa + kb + 128, 0, 0);   // stream x ahead

        // ---- A fragment: 16 f32 -> bf16 hi + bf16 residual lo ----
        // lane's K order: {k0..k0+15}, k0 = kb + half*16
        const float4* a4 = (const float4*)(pa + kb);
        float4 A0 = a4[0], A1 = a4[1], A2 = a4[2], A3 = a4[3];
        float af[16] = {A0.x,A0.y,A0.z,A0.w, A1.x,A1.y,A1.z,A1.w,
                        A2.x,A2.y,A2.z,A2.w, A3.x,A3.y,A3.z,A3.w};
        v16bf ahi, alo;
        #pragma unroll
        for (int i = 0; i < 16; ++i) {
            __bf16 h = (__bf16)af[i];            // v_cvt_pk_bf16_f32 (RNE)
            ahi[i] = h;
            alo[i] = (__bf16)(af[i] - (float)h); // residual
        }

        // ---- 4 expert tiles of B; 3 WMMAs each (hi*hi + lo*hi + hi*lo) ----
        #pragma unroll
        for (int t = 0; t < 4; ++t) {
            const int e = t * 16 + col;
            v16bf bhi, blo;
            if (PRECONV) {
                // B fragment in this code's contiguous K order (same as A): lane holds expert column e;
                // lanes<16: K = kb+0..15, lanes>=16: K = kb+16..31
                // -> one aligned 32-byte v16bf per precision.
                const size_t idx = (size_t)e * (DIM / 16) + (kb >> 4) + half;
                bhi = whi[idx];
                blo = wlo[idx];
            } else {
                const float4* b4 =
                    (const float4*)(w + (size_t)e * DIM + kb + half * 16);
                float4 B0 = b4[0], B1 = b4[1], B2 = b4[2], B3 = b4[3];
                float bf[16] = {B0.x,B0.y,B0.z,B0.w, B1.x,B1.y,B1.z,B1.w,
                                B2.x,B2.y,B2.z,B2.w, B3.x,B3.y,B3.z,B3.w};
                #pragma unroll
                for (int i = 0; i < 16; ++i) {
                    __bf16 h = (__bf16)bf[i];
                    bhi[i] = h;
                    blo[i] = (__bf16)(bf[i] - (float)h);
                }
            }

            acc[t] = __builtin_amdgcn_wmma_f32_16x16x32_bf16(
                         false, ahi, false, bhi, (short)0, acc[t], false, false);
            acc[t] = __builtin_amdgcn_wmma_f32_16x16x32_bf16(
                         false, alo, false, bhi, (short)0, acc[t], false, false);
            acc[t] = __builtin_amdgcn_wmma_f32_16x16x32_bf16(
                         false, ahi, false, blo, (short)0, acc[t], false, false);
        }
    }

    // ---- sigmoid + scatter to LDS ----
    // C layout: VGPR r -> token M = r + 8*half (lane group), N = col.
    #pragma unroll
    for (int t = 0; t < 4; ++t) {
        #pragma unroll
        for (int r = 0; r < 8; ++r) {
            float v = acc[t][r];
            float s = 1.0f / (1.0f + __expf(-v));
            sc[wave * 16 + r + 8 * half][t * 16 + col] = s;
        }
    }
    __syncthreads();

    // ---- top-8 per token: 128 threads, one token each ----
    if (tid < TOK_PER_BLOCK) {
        const int tok = tokBase + tid;
        const float* row = sc[tid];
        unsigned long long picked = 0ull;
        float vals[TOPK];
        int   idxs[TOPK];
        float sum = 0.0f;
        #pragma unroll
        for (int j = 0; j < TOPK; ++j) {
            float best = -1.0f;   // sigmoid outputs are in (0,1)
            int   bi   = 0;
            for (int i = 0; i < NEXP; ++i) {
                float v = row[i];
                bool free_slot = ((picked >> i) & 1ull) == 0ull;
                if (free_slot && v > best) { best = v; bi = i; }
            }
            picked |= (1ull << bi);
            vals[j] = best;
            idxs[j] = bi;
            sum += best;
        }
        float inv = 1.0f / sum;   // ROUTE_SCALE == 1.0
        #pragma unroll
        for (int j = 0; j < TOPK; ++j) {
            out[(size_t)tok * TOPK + j] = vals[j] * inv;
            out[(size_t)TOKENS * TOPK + (size_t)tok * TOPK + j] = (float)idxs[j];
        }
    }
}

extern "C" void kernel_launch(void* const* d_in, const int* in_sizes, int n_in,
                              void* d_out, int out_size, void* d_ws, size_t ws_size,
                              hipStream_t stream) {
    (void)in_sizes; (void)n_in; (void)out_size;
    const float* x = (const float*)d_in[0];
    const float* w = (const float*)d_in[1];
    float* out = (float*)d_out;

    const dim3 grid(TOKENS / TOK_PER_BLOCK);   // 128 blocks
    const dim3 block(256);                      // 8 wave32 waves

    const size_t need = (size_t)NEXP * DIM * sizeof(unsigned short) * 2; // 1 MB
    if (d_ws != nullptr && ws_size >= need) {
        unsigned int* whi_u = (unsigned int*)d_ws;
        unsigned int* wlo_u = whi_u + (NEXP * DIM / 2);   // 512 KB offset
        // 262144 weight elems / 4 per thread = 65536 threads = 256 blocks
        hipLaunchKernelGGL(conv_weight, dim3(256), dim3(256), 0, stream,
                           w, whi_u, wlo_u);
        hipLaunchKernelGGL((moe_gate_wmma<true>), grid, block, 0, stream,
                           x, w, (const v16bf*)whi_u, (const v16bf*)wlo_u, out);
    } else {
        hipLaunchKernelGGL((moe_gate_wmma<false>), grid, block, 0, stream,
                           x, w, (const v16bf*)nullptr, (const v16bf*)nullptr, out);
    }
}
